// Fire_model_hybrid_17291538333714
// MI455X (gfx1250) — matrix operand order repaired and checked by reading, not compiled, not run on hardware
//
#include <hip/hip_runtime.h>
#include <hip/hip_bf16.h>

// ---- problem constants -----------------------------------------------------
#define B_SZ   512
#define T_SZ   64
#define IN_DIM 21
#define NU     128
#define D_SZ   19
#define DN     (D_SZ*NU)          // 2432
#define KIN    (D_SZ*(NU+1))      // 2451
#define KPAD   2464               // 77*32, padded K for gate GEMM
#define KT_G   77                 // gate K tiles of 32
#define NTPG   152                // 2432/16 N-tiles per gate
#define NT_TOT 456                // 3 gates * 152
#define MT     32                 // 512/16 M-tiles
#define GW     6                  // waves per gate-GEMM workgroup (6*4 tiles * 19 blk = 456)

typedef __attribute__((ext_vector_type(16))) _Float16 v16h;
typedef __attribute__((ext_vector_type(8)))  _Float16 v8h;
typedef __attribute__((ext_vector_type(8)))  float    v8f;

__device__ __forceinline__ float sigmoidf_(float v){ return 1.0f/(1.0f+expf(-v)); }

// Build a 16-half A fragment from two row-major 8-half chunks (k0, k0+16).
__device__ __forceinline__ v16h make_afrag(const _Float16* src)
{
  v8h lo = *(const v8h*)src;
  v8h hi = *(const v8h*)(src + 16);
  v16h a;
#pragma unroll
  for (int i = 0; i < 8; ++i) { a[i] = lo[i]; a[i+8] = hi[i]; }
  return a;
}

// ---- one-time weight repack into WMMA B-operand lane layout ----------------
// B tile (32K x 16N) per wave: lane = (n&15) + 16*((k>>3)&1), half idx = (k&7) + 8*(k>=16).
// Packed layout: [nt][kt][512 halfs], nt in [0,456) covering gates i,f,o.
// Column permutation: inp is re-ordered to [h(2432) | x(19) | zeros], so
// k<2432 -> original col 19+k, 2432<=k<2451 -> col k-2432, else zero pad.
__global__ void pack_gates(const float* __restrict__ Wi, const float* __restrict__ Wf,
                           const float* __restrict__ Wo, _Float16* __restrict__ Wg)
{
  size_t p = (size_t)blockIdx.x*256 + threadIdx.x;       // < 456*77*512
  int    r    = (int)(p & 511);
  size_t tile = p >> 9;
  int kt   = (int)(tile % KT_G);
  int nt   = (int)(tile / KT_G);
  int gate = nt / NTPG;
  int lane = r >> 4, idx = r & 15;
  int n = (nt % NTPG)*16 + (lane & 15);
  int k = kt*32 + (idx & 7) + ((lane >> 4) << 3) + ((idx >> 3) << 4);
  float v = 0.0f;
  if (k < KIN) {
    int col = (k < DN) ? (19 + k) : (k - DN);
    const float* W = (gate == 0) ? Wi : ((gate == 1) ? Wf : Wo);
    v = W[(size_t)n*KIN + col];
  }
  Wg[p] = (_Float16)v;
}

// W_j[d] is [K=128][N=128]; packed [d][nt(8)][kt(4)][512]
__global__ void pack_wj(const float* __restrict__ Wj, _Float16* __restrict__ Wp)
{
  size_t p = (size_t)blockIdx.x*256 + threadIdx.x;       // < 19*8*4*512
  int    r    = (int)(p & 511);
  size_t tile = p >> 9;
  int kt = (int)(tile & 3);
  int nt = (int)((tile >> 2) & 7);
  int d  = (int)(tile >> 5);
  int lane = r >> 4, idx = r & 15;
  int n = nt*16 + (lane & 15);
  int k = kt*32 + (idx & 7) + ((lane >> 4) << 3) + ((idx >> 3) << 4);
  Wp[p] = (_Float16)Wj[((size_t)d*NU + k)*NU + n];
}

// ---- init: c = 0, inp = [h=0 | x_FF[:,0,:] | pad=0] ------------------------
__global__ void init_state(const float* __restrict__ x, _Float16* __restrict__ inp,
                           float* __restrict__ c)
{
  int i = blockIdx.x*256 + threadIdx.x;                  // < 512*2464
  int b = i / KPAD, col = i % KPAD;
  float v = 0.0f;
  if (col >= DN && col < KIN) {
    int d = col - DN;
    v = x[((size_t)b*T_SZ)*IN_DIM + (d < 16 ? d : d + 2)];
  }
  inp[i] = (_Float16)v;
  if (col < DN) c[(size_t)b*DN + col] = 0.0f;
}

// ---- fused i/f/o gate GEMM: pre[g][512][2432] = inp @ Wgᵀ ------------------
// Workgroup = 6 waves sharing one 16-row A panel staged in LDS (pre-swizzled
// into WMMA A-fragment order so the inner loop is one 32B DS read per lane).
// Each wave owns a 16M x 64N tile quad; B is double-buffered from global/L2.
#define LOADB4(dst, bb, kt)                                        \
  do {                                                             \
    const _Float16* _bp = (bb) + (size_t)(kt)*512;                 \
    dst[0] = *(const v16h*)(_bp);                                  \
    dst[1] = *(const v16h*)(_bp + (size_t)1*KT_G*512);             \
    dst[2] = *(const v16h*)(_bp + (size_t)2*KT_G*512);             \
    dst[3] = *(const v16h*)(_bp + (size_t)3*KT_G*512);             \
  } while (0)

#define WMMA4(acc, a, bv)                                                      \
  do {                                                                         \
    acc[0] = __builtin_amdgcn_wmma_f32_16x16x32_f16(false, a, false, bv[0],    \
                                                    (short)0, acc[0], false, false); \
    acc[1] = __builtin_amdgcn_wmma_f32_16x16x32_f16(false, a, false, bv[1],    \
                                                    (short)0, acc[1], false, false); \
    acc[2] = __builtin_amdgcn_wmma_f32_16x16x32_f16(false, a, false, bv[2],    \
                                                    (short)0, acc[2], false, false); \
    acc[3] = __builtin_amdgcn_wmma_f32_16x16x32_f16(false, a, false, bv[3],    \
                                                    (short)0, acc[3], false, false); \
  } while (0)

__global__ __launch_bounds__(GW*32) void gate_gemm(const _Float16* __restrict__ inp,
                                                   const _Float16* __restrict__ Wg,
                                                   float* __restrict__ pre)
{
  extern __shared__ _Float16 As[];                       // [KT_G][32 lanes][16 halfs]
  int tid  = threadIdx.x;
  int wave = tid >> 5, lane = tid & 31;
  int mt   = blockIdx.x;

  // ---- cooperative A staging: global row-major -> LDS fragment layout ----
  for (int i = tid; i < KT_G*32; i += GW*32) {
    int kt = i >> 5, l = i & 31;
    int row = l & 15, k0 = (l >> 4) * 8;
    const _Float16* src = inp + (size_t)(mt*16 + row)*KPAD + kt*32 + k0;
    *(v16h*)(As + (size_t)i*16) = make_afrag(src);
  }
  __syncthreads();

  int nq = blockIdx.y*GW + wave;                         // [0,114)
  const _Float16* bbase = Wg + (size_t)(nq*4)*KT_G*512 + lane*16;
  const _Float16* afrag = As + (size_t)lane*16;

  v8f  acc[4] = {};
  v16h b0[4], b1[4];
  v16h a;
  LOADB4(b0, bbase, 0);
  for (int kt = 0; kt + 1 < KT_G; kt += 2) {
    __builtin_prefetch(bbase + (size_t)(kt + 3)*512, 0, 1);  // global_prefetch_b8
    LOADB4(b1, bbase, kt + 1);
    a = *(const v16h*)(afrag + (size_t)kt*512);
    WMMA4(acc, a, b0);
    if (kt + 2 < KT_G) LOADB4(b0, bbase, kt + 2);
    a = *(const v16h*)(afrag + (size_t)(kt + 1)*512);
    WMMA4(acc, a, b1);
  }
  a = *(const v16h*)(afrag + (size_t)(KT_G - 1)*512);
  WMMA4(acc, a, b0);                                     // kt = 76

  int mbase = mt*16 + 8*(lane >> 4);
  int nloc  = lane & 15;
#pragma unroll
  for (int q = 0; q < 4; ++q) {
    int nt   = nq*4 + q;
    int gate = nt / NTPG;
    int ncol = (nt % NTPG)*16 + nloc;
    float* op = pre + ((size_t)gate*B_SZ + mbase)*DN + ncol;
#pragma unroll
    for (int rr = 0; rr < 8; ++rr) op[(size_t)rr*DN] = acc[q][rr];
  }
}

// ---- per-variable candidate GEMM: jpre[b, d*128+n] = h[:,d,:] @ W_j[d] -----
// One wave computes 16M x 128N for one variable d; software-pipelined over
// the 4 K-tiles with ping-pong A/B register buffers.
#define LOADB8(dst, bb, kt)                                        \
  do {                                                             \
    _Pragma("unroll")                                              \
    for (int _n = 0; _n < 8; ++_n)                                 \
      dst[_n] = *(const v16h*)((bb) + ((size_t)_n*4 + (kt))*512);  \
  } while (0)

#define WMMA8(acc, a, bv)                                                      \
  do {                                                                         \
    _Pragma("unroll")                                                          \
    for (int _n = 0; _n < 8; ++_n)                                             \
      acc[_n] = __builtin_amdgcn_wmma_f32_16x16x32_f16(false, a, false, bv[_n],\
                                                 (short)0, acc[_n], false, false); \
  } while (0)

__global__ __launch_bounds__(32) void j_gemm(const _Float16* __restrict__ inp,
                                             const _Float16* __restrict__ Wp,
                                             float* __restrict__ jpre)
{
  int lane = threadIdx.x;
  int mt = blockIdx.x, d = blockIdx.y;
  int m  = mt*16 + (lane & 15);
  int k0 = (lane >> 4) * 8;
  const _Float16* arow  = inp + (size_t)m*KPAD + d*NU;   // h lives at offset 0
  const _Float16* bbase = Wp + (size_t)d*32*512 + lane*16;

  v8f  acc[8] = {};
  v16h a0, a1, bb0[8], bb1[8];
  a0 = make_afrag(arow + 0*32 + k0);  LOADB8(bb0, bbase, 0);
  a1 = make_afrag(arow + 1*32 + k0);  LOADB8(bb1, bbase, 1);
  WMMA8(acc, a0, bb0);                                   // kt 0
  a0 = make_afrag(arow + 2*32 + k0);  LOADB8(bb0, bbase, 2);
  WMMA8(acc, a1, bb1);                                   // kt 1
  a1 = make_afrag(arow + 3*32 + k0);  LOADB8(bb1, bbase, 3);
  WMMA8(acc, a0, bb0);                                   // kt 2
  WMMA8(acc, a1, bb1);                                   // kt 3

  int mbase = mt*16 + 8*(lane >> 4);
  int nloc  = lane & 15;
#pragma unroll
  for (int nt = 0; nt < 8; ++nt) {
    float* op = jpre + (size_t)mbase*DN + d*NU + nt*16 + nloc;
#pragma unroll
    for (int rr = 0; rr < 8; ++rr) op[(size_t)rr*DN] = acc[nt][rr];
  }
}

// ---- elementwise LSTM update ------------------------------------------------
__global__ void update_step(const float* __restrict__ x,  const float* __restrict__ Uj,
                            const float* __restrict__ bj, const float* __restrict__ Wib,
                            const float* __restrict__ Wfb,const float* __restrict__ Wob,
                            const float* __restrict__ pre,const float* __restrict__ jpre,
                            float* __restrict__ c, _Float16* __restrict__ inp,
                            _Float16* __restrict__ outs, float* __restrict__ hfin, int t)
{
  int idx = blockIdx.x*256 + threadIdx.x;                // < 512*2432
  int b = idx / DN, dn = idx % DN;
  int d = dn >> 7, n = dn & 127;
  const size_t S = (size_t)B_SZ*DN;
  float iv = sigmoidf_(pre[idx]       + Wib[dn]);
  float fv = sigmoidf_(pre[S + idx]   + Wfb[dn]);
  float ov = sigmoidf_(pre[2*S + idx] + Wob[dn]);
  float xt = x[((size_t)b*T_SZ + t)*IN_DIM + (d < 16 ? d : d + 2)];
  float jv = tanhf(jpre[idx] + xt*Uj[d*NU + n] + bj[dn]);
  float cn = c[idx]*fv + iv*jv;
  float hv = ov*tanhf(cn);
  c[idx]    = cn;
  hfin[idx] = hv;
  outs[((size_t)(b*T_SZ + t))*DN + dn] = (_Float16)hv;
  inp[(size_t)b*KPAD + dn] = (_Float16)hv;               // h region (offset 0)
  if (dn < D_SZ && t + 1 < T_SZ) {                       // stage x_{t+1}
    inp[(size_t)b*KPAD + DN + dn] =
      (_Float16)x[((size_t)b*T_SZ + t + 1)*IN_DIM + (dn < 16 ? dn : dn + 2)];
  }
}

// ---- temporal attention: g_n[b,d,:] = sum_t softmaxish(t) * out[b,t,d,:] ---
__global__ __launch_bounds__(128) void attention(const _Float16* __restrict__ outs,
                                                 const float* __restrict__ Fa,
                                                 const float* __restrict__ Fab,
                                                 float* __restrict__ gn)
{
  int b = blockIdx.x, d = blockIdx.y, n = threadIdx.x;
  __shared__ float red[128];
  float fa   = Fa[d*NU + n];
  float bias = Fab[d];
  float acc = 0.0f, denom = 0.0f;
  for (int t = 0; t < T_SZ; ++t) {
    float v = (float)outs[((size_t)(b*T_SZ + t))*DN + d*NU + n];
    red[n] = v*fa;
    __syncthreads();
    for (int s = 64; s > 0; s >>= 1) { if (n < s) red[n] += red[n + s]; __syncthreads(); }
    float w = expf(tanhf(red[0] + bias));
    acc += w*v; denom += w;
    __syncthreads();
  }
  gn[((size_t)(b*D_SZ + d))*NU + n] = acc/denom;
}

// ---- final head: mu/beta over hg=[g_n,h_fin], softmax over D, gates --------
__global__ __launch_bounds__(256) void final_head(const float* __restrict__ gn,
    const float* __restrict__ hfin, const float* __restrict__ Phiw,
    const float* __restrict__ Phib, const float* __restrict__ Fbw,
    const float* __restrict__ Fbb,  const float* __restrict__ faw,
    const float* __restrict__ fab,  const float* __restrict__ hsw,
    const float* __restrict__ hsb,  const float* __restrict__ x,
    float* __restrict__ out)
{
  int b = blockIdx.x, k = threadIdx.x;                   // k in [0,256)
  __shared__ float rm[256], rb[256];
  float summ = 0.0f, sumb = 0.0f;
  for (int d = 0; d < D_SZ; ++d) {
    float v = (k < NU) ? gn[((size_t)(b*D_SZ + d))*NU + k]
                       : hfin[(size_t)b*DN + d*NU + (k - NU)];
    rm[k] = v*Phiw[k];
    rb[k] = v*Fbw[k];
    __syncthreads();
    for (int s = 128; s > 0; s >>= 1) {
      if (k < s) { rm[k] += rm[k + s]; rb[k] += rb[k + s]; }
      __syncthreads();
    }
    float mu  = rm[0] + Phib[0];
    float bet = expf(tanhf(rb[0] + Fbb[0]));
    summ += bet*mu; sumb += bet;
    __syncthreads();
  }
  if (k == 0) {
    float fFF   = sigmoidf_(summ/sumb);
    float fnpp  = sigmoidf_(x[((size_t)b*T_SZ + T_SZ - 1)*IN_DIM + 0]*faw[0] + fab[0]);
    float fpopu = sigmoidf_(x[((size_t)b*T_SZ + T_SZ - 1)*IN_DIM + 16]*hsw[0] + hsb[0]);
    out[b] = fFF*fnpp*fpopu;
  }
}

// ---- host launcher ---------------------------------------------------------
extern "C" void kernel_launch(void* const* d_in, const int* in_sizes, int n_in,
                              void* d_out, int out_size, void* d_ws, size_t ws_size,
                              hipStream_t stream)
{
  (void)in_sizes; (void)n_in; (void)out_size; (void)ws_size;
  const float* x    = (const float*)d_in[0];
  const float* Uj   = (const float*)d_in[1];
  const float* Wj   = (const float*)d_in[2];
  const float* bj   = (const float*)d_in[3];
  const float* Wi   = (const float*)d_in[4];
  const float* Wib  = (const float*)d_in[5];
  const float* Wf   = (const float*)d_in[6];
  const float* Wfb  = (const float*)d_in[7];
  const float* Wo   = (const float*)d_in[8];
  const float* Wob  = (const float*)d_in[9];
  const float* Fa   = (const float*)d_in[10];
  const float* Fab  = (const float*)d_in[11];
  const float* Fbw  = (const float*)d_in[12];
  const float* Fbb  = (const float*)d_in[13];
  const float* Phiw = (const float*)d_in[14];
  const float* Phib = (const float*)d_in[15];
  const float* faw  = (const float*)d_in[16];
  const float* fab  = (const float*)d_in[17];
  const float* hsw  = (const float*)d_in[18];
  const float* hsb  = (const float*)d_in[19];
  float* out = (float*)d_out;

  char* ws = (char*)d_ws;
  size_t off = 0;
  auto take = [&](size_t bytes) -> char* {
    char* p = ws + off;
    off = (off + bytes + 255) & ~(size_t)255;
    return p;
  };
  _Float16* Wg   = (_Float16*)take((size_t)NT_TOT*KT_G*512*2);   // ~36 MB
  _Float16* Wjp  = (_Float16*)take((size_t)D_SZ*32*512*2);       // ~0.6 MB
  _Float16* inp  = (_Float16*)take((size_t)B_SZ*KPAD*2);         // ~2.5 MB
  float*    c    = (float*)   take((size_t)B_SZ*DN*4);           // ~5 MB
  float*    pre  = (float*)   take((size_t)3*B_SZ*DN*4);         // ~15 MB
  float*    jpre = (float*)   take((size_t)B_SZ*DN*4);           // ~5 MB
  float*    hfin = (float*)   take((size_t)B_SZ*DN*4);           // ~5 MB
  float*    gn   = (float*)   take((size_t)B_SZ*D_SZ*NU*4);      // ~5 MB
  _Float16* outs = (_Float16*)take((size_t)B_SZ*T_SZ*DN*2);      // ~159 MB

  pack_gates<<<dim3((NT_TOT*KT_G*512)/256), 256, 0, stream>>>(Wi, Wf, Wo, Wg);
  pack_wj   <<<dim3((D_SZ*32*512)/256),     256, 0, stream>>>(Wj, Wjp);
  init_state<<<dim3((B_SZ*KPAD)/256),       256, 0, stream>>>(x, inp, c);

  const size_t lds_bytes = (size_t)KT_G*32*16*sizeof(_Float16);  // 78,848 B
  for (int t = 0; t < T_SZ; ++t) {
    gate_gemm<<<dim3(MT, (NT_TOT/4)/GW), GW*32, lds_bytes, stream>>>(inp, Wg, pre);
    j_gemm   <<<dim3(MT, D_SZ),          32,    0,         stream>>>(inp, Wjp, jpre);
    update_step<<<dim3((B_SZ*DN)/256), 256, 0, stream>>>(x, Uj, bj, Wib, Wfb, Wob,
                                                         pre, jpre, c, inp, outs, hfin, t);
  }

  attention<<<dim3(B_SZ, D_SZ), 128, 0, stream>>>(outs, Fa, Fab, gn);
  final_head<<<dim3(B_SZ), 256, 0, stream>>>(gn, hfin, Phiw, Phib, Fbw, Fbb,
                                             faw, fab, hsw, hsb, x, out);
}
